// GATWithMultipleLinearLayers_88433376624937
// MI455X (gfx1250) — matrix operand order repaired and checked by reading, not compiled, not run on hardware
//
#include <hip/hip_runtime.h>
#include <hip/hip_bf16.h>
#include <stdint.h>

typedef __bf16 bf16;
typedef __attribute__((ext_vector_type(16))) __bf16 v16bf;
typedef __attribute__((ext_vector_type(8)))  float  v8f;

union Frag16 { v16bf v; unsigned int u[8]; uint4 q[2]; };

#define LRELU_SLOPE 0.2f

// ---------------------------------------------------------------------------
// f32 -> bf16 convert (grid-stride)
// ---------------------------------------------------------------------------
__global__ void k_cvt_bf16(const float* __restrict__ in, bf16* __restrict__ out, int n) {
    int i = blockIdx.x * blockDim.x + threadIdx.x;
    int stride = gridDim.x * blockDim.x;
    for (; i < n; i += stride) out[i] = (bf16)in[i];
}

// f32 [K][N] -> bf16 transposed [N][K]  (weights; tiny, done once)
__global__ void k_cvt_bf16_t(const float* __restrict__ in, bf16* __restrict__ out, int K, int N) {
    int i = blockIdx.x * blockDim.x + threadIdx.x;
    int stride = gridDim.x * blockDim.x;
    int total = K * N;
    for (; i < total; i += stride) {
        int k = i / N, n = i - k * N;
        out[(size_t)n * K + k] = (bf16)in[i];
    }
}

// f32 -> bf16 with optional relu
__global__ void k_relu_cvt(const float* __restrict__ in, bf16* __restrict__ out, int n, int do_relu) {
    int i = blockIdx.x * blockDim.x + threadIdx.x;
    int stride = gridDim.x * blockDim.x;
    for (; i < n; i += stride) {
        float v = in[i];
        if (do_relu) v = fmaxf(v, 0.f);
        out[i] = (bf16)v;
    }
}

// ---------------------------------------------------------------------------
// bf16 WMMA GEMM:  C[M,N] = A[Mp,K] @ Bt[N,K]^T  (+bias0 +bias1, optional relu)
//  - Block: 256 threads = 8 waves (2 M x 4 N); block tile 64(M) x 128(N)
//  - Each wave: 32x32 via 2x2 v_wmma_f32_16x16x32_bf16 with fragment reuse
//  - A tile double-buffered in LDS via async global->LDS copies (no VGPR hop)
//  - B fragments loaded straight from global (pre-transposed weights, L2-hot)
//  - A rows are padded to a multiple of 64 by the caller: no load guards
// ---------------------------------------------------------------------------
__global__ __launch_bounds__(256)
void k_gemm_bf16(const bf16* __restrict__ A, const bf16* __restrict__ Bt,
                 const float* __restrict__ bias0, const float* __restrict__ bias1,
                 float* __restrict__ Cf, bf16* __restrict__ Cb,
                 int M, int N, int K, int do_relu)
{
    __shared__ __align__(16) bf16 As[2][64][40];   // [buf][m][k], +8 pad (16B-aligned rows)

    const int tid  = threadIdx.x;
    const int lane = tid & 31;
    const int wave = tid >> 5;
    const int wR   = wave >> 2;       // 0..1 : 32-row subtile
    const int wC   = wave & 3;        // 0..3 : 32-col subtile
    const int mBase = blockIdx.y * 64;
    const int nBase = blockIdx.x * 128;

    // cooperative A copy: one 16B async copy per thread per K-step
    const int aR = tid >> 2;          // 0..63
    const int aC = (tid & 3) * 8;     // 0,8,16,24
    const bf16* aSrc = A + (size_t)(mBase + aR) * K + aC;

    // fragment index math (ISA 7.12.2 wave32 layouts)
    const int l15 = lane & 15;
    const int rA0 = wR * 32 + l15;               // A rows, subtile 0 (subtile 1 = +16)
    const int kA  = (lane < 16) ? 0 : 8;         // A: lanes 0-15 K{0-7,16-23}; 16-31 K{8-15,24-31}
    const int kB  = (lane < 16) ? 0 : 8;         // B: lanes 0-15 K{0-7,16-23}; 16-31 K{8-15,24-31}
    const bf16* bRow0 = Bt + (size_t)(nBase + wC * 32 + l15) * K + kB;
    const bf16* bRow1 = bRow0 + (size_t)16 * K;

    v8f acc00 = {}, acc01 = {}, acc10 = {}, acc11 = {};

    const int S = K >> 5;

    // prologue: tile 0 -> buf 0 (async copy, no VGPR round trip)
    {
        unsigned     ldsA = (unsigned)(uintptr_t)&As[0][aR][aC];
        unsigned long long ga = (unsigned long long)(uintptr_t)aSrc;
        asm volatile("global_load_async_to_lds_b128 %0, %1, off" :: "v"(ldsA), "v"(ga) : "memory");
        asm volatile("s_wait_asynccnt 0" ::: "memory");
    }
    __syncthreads();

    for (int s = 0; s < S; ++s) {
        const int k0  = s << 5;
        const int buf = s & 1;

        // prefetch next A tile into the other buffer (async; wait just before barrier)
        if (s + 1 < S) {
            unsigned     ldsA = (unsigned)(uintptr_t)&As[buf ^ 1][aR][aC];
            unsigned long long ga = (unsigned long long)(uintptr_t)(aSrc + k0 + 32);
            asm volatile("global_load_async_to_lds_b128 %0, %1, off" :: "v"(ldsA), "v"(ga) : "memory");
        }

        // B fragments: two 8-bf16 K runs (kB, kB+16) per lane -> 2x b128 each
        Frag16 bf0, bf1;
        bf0.q[0] = *(const uint4*)(bRow0 + k0);
        bf0.q[1] = *(const uint4*)(bRow0 + k0 + 16);
        bf1.q[0] = *(const uint4*)(bRow1 + k0);
        bf1.q[1] = *(const uint4*)(bRow1 + k0 + 16);

        // A fragments from LDS (vectorizes to ds_load_2addr_b64)
        Frag16 af0, af1;
        #pragma unroll
        for (int j = 0; j < 4; ++j) {
            af0.u[j]     = *(const unsigned int*)(&As[buf][rA0][kA + 2 * j]);
            af0.u[4 + j] = *(const unsigned int*)(&As[buf][rA0][kA + 16 + 2 * j]);
            af1.u[j]     = *(const unsigned int*)(&As[buf][rA0 + 16][kA + 2 * j]);
            af1.u[4 + j] = *(const unsigned int*)(&As[buf][rA0 + 16][kA + 16 + 2 * j]);
        }

        acc00 = __builtin_amdgcn_wmma_f32_16x16x32_bf16(false, af0.v, false, bf0.v, (short)0, acc00, false, false);
        acc01 = __builtin_amdgcn_wmma_f32_16x16x32_bf16(false, af0.v, false, bf1.v, (short)0, acc01, false, false);
        acc10 = __builtin_amdgcn_wmma_f32_16x16x32_bf16(false, af1.v, false, bf0.v, (short)0, acc10, false, false);
        acc11 = __builtin_amdgcn_wmma_f32_16x16x32_bf16(false, af1.v, false, bf1.v, (short)0, acc11, false, false);

        if (s + 1 < S) asm volatile("s_wait_asynccnt 0" ::: "memory");
        __syncthreads();
    }

    // epilogue: C layout: VGPR v -> row v (lanes 0-15) / v+8 (lanes 16-31), col = lane%16
    const int colA   = nBase + wC * 32 + l15;
    const int rowOff = (lane < 16) ? 0 : 8;
    auto store_tile = [&](const v8f& a, int rBase, int col) {
        float badd = 0.f;
        if (bias0) badd += bias0[col];
        if (bias1) badd += bias1[col];
        #pragma unroll
        for (int v = 0; v < 8; ++v) {
            const int row = rBase + rowOff + v;
            if (row < M) {
                float val = a[v] + badd;
                if (do_relu) val = fmaxf(val, 0.f);
                if (Cf) Cf[(size_t)row * N + col] = val;
                if (Cb) Cb[(size_t)row * N + col] = (bf16)val;
            }
        }
    };
    store_tile(acc00, mBase + wR * 32,      colA);
    store_tile(acc01, mBase + wR * 32,      colA + 16);
    store_tile(acc10, mBase + wR * 32 + 16, colA);
    store_tile(acc11, mBase + wR * 32 + 16, colA + 16);
}

// ---------------------------------------------------------------------------
// attention dots: a_s[r] = <h[r,:], att_s>, a_d[r] = <h[r,:], att_d>
// ---------------------------------------------------------------------------
__global__ void k_attn_dots(const bf16* __restrict__ h,
                            const float* __restrict__ att_s, const float* __restrict__ att_d,
                            float* __restrict__ a_s, float* __restrict__ a_d, int M, int H) {
    int gw = (blockIdx.x * blockDim.x + threadIdx.x) >> 5;
    int lane = threadIdx.x & 31;
    if (gw >= M) return;
    const bf16* hr = h + (size_t)gw * H;
    float ss = 0.f, sd = 0.f;
    for (int c = lane; c < H; c += 32) {
        float v = (float)hr[c];
        ss += v * att_s[c];
        sd += v * att_d[c];
    }
    #pragma unroll
    for (int off = 16; off; off >>= 1) {
        ss += __shfl_down(ss, off);
        sd += __shfl_down(sd, off);
    }
    if (lane == 0) { a_s[gw] = ss; a_d[gw] = sd; }
}

// ---------------------------------------------------------------------------
// edge-softmax pieces
// ---------------------------------------------------------------------------
__global__ void k_init_ms(float* __restrict__ m, float* __restrict__ s, int n) {
    int i = blockIdx.x * blockDim.x + threadIdx.x;
    if (i < n) { m[i] = -INFINITY; s[i] = 0.f; }
}

__device__ inline void atomicMaxF(float* addr, float v) {
    if (v >= 0.f) atomicMax((int*)addr, __float_as_int(v));
    else          atomicMin((unsigned int*)addr, __float_as_uint(v));
}

__device__ inline void atomAddF(float* p, float v) { unsafeAtomicAdd(p, v); }

__global__ void k_edge_logits_max(const int* __restrict__ ei,
                                  const float* __restrict__ a_s, const float* __restrict__ a_d,
                                  float* __restrict__ logit, float* __restrict__ m,
                                  int E, int M) {
    int e = blockIdx.x * blockDim.x + threadIdx.x;
    if (e >= E) return;
    int s_ = min(max(ei[e], 0), M - 1);
    int d_ = min(max(ei[E + e], 0), M - 1);
    float l = a_s[s_] + a_d[d_];
    l = (l > 0.f) ? l : LRELU_SLOPE * l;
    logit[e] = l;
    atomicMaxF(&m[d_], l);
}

__global__ void k_edge_exp_sum(const int* __restrict__ ei,
                               const float* __restrict__ logit, const float* __restrict__ m,
                               float* __restrict__ ew, float* __restrict__ s,
                               int E, int M) {
    int e = blockIdx.x * blockDim.x + threadIdx.x;
    if (e >= E) return;
    int d_ = min(max(ei[E + e], 0), M - 1);
    float v = __expf(logit[e] - m[d_]);
    ew[e] = v;
    atomAddF(&s[d_], v);
}

// one wave per edge: agg[dst,:] += h[src,:] * alpha   (agg pre-seeded by GEMM)
__global__ void k_edge_aggregate(const int* __restrict__ ei,
                                 const float* __restrict__ ew, const float* __restrict__ s,
                                 const bf16* __restrict__ h, float* __restrict__ agg,
                                 int E, int M, int H) {
    int t = blockIdx.x * blockDim.x + threadIdx.x;
    int e = t >> 5;
    if (e >= E) return;
    int lane = t & 31;
    int s_ = min(max(ei[e], 0), M - 1);
    int d_ = min(max(ei[E + e], 0), M - 1);
    float alpha = ew[e] / (s[d_] + 1e-16f);
    const bf16* hs = h + (size_t)s_ * H;
    float* od = agg + (size_t)d_ * H;
    for (int c = lane; c < H; c += 32)
        atomAddF(&od[c], (float)hs[c] * alpha);
}

// ---------------------------------------------------------------------------
// classifier + row softmax: one wave per row; H2=256, 5 classes
// ---------------------------------------------------------------------------
__global__ void k_classify(const bf16* __restrict__ h, const float* __restrict__ fcw,
                           const float* __restrict__ fcb, float* __restrict__ out, int M) {
    int gw = (blockIdx.x * blockDim.x + threadIdx.x) >> 5;
    int lane = threadIdx.x & 31;
    if (gw >= M) return;
    const bf16* hr = h + (size_t)gw * 256;
    float hv[8];
    #pragma unroll
    for (int j = 0; j < 8; ++j) hv[j] = (float)hr[lane + 32 * j];
    float logits[5];
    #pragma unroll
    for (int c = 0; c < 5; ++c) {
        float p = 0.f;
        #pragma unroll
        for (int j = 0; j < 8; ++j) p += hv[j] * fcw[(lane + 32 * j) * 5 + c];
        #pragma unroll
        for (int off = 16; off; off >>= 1) p += __shfl_down(p, off);
        logits[c] = p + fcb[c];
    }
    if (lane == 0) {
        float mx = logits[0];
        #pragma unroll
        for (int c = 1; c < 5; ++c) mx = fmaxf(mx, logits[c]);
        float ex[5], sum = 0.f;
        #pragma unroll
        for (int c = 0; c < 5; ++c) { ex[c] = __expf(logits[c] - mx); sum += ex[c]; }
        float inv = 1.f / sum;
        float* o = out + (size_t)gw * 5;
        #pragma unroll
        for (int c = 0; c < 5; ++c) o[c] = ex[c] * inv;
    }
}

// ---------------------------------------------------------------------------
extern "C" void kernel_launch(void* const* d_in, const int* in_sizes, int n_in,
                              void* d_out, int out_size, void* d_ws, size_t ws_size,
                              hipStream_t stream) {
    (void)n_in; (void)out_size; (void)ws_size;

    const float* x   = (const float*)d_in[0];
    const int*   ei  = (const int*)d_in[1];
    const float* W1  = (const float*)d_in[2];
    const float* as1 = (const float*)d_in[3];
    const float* ad1 = (const float*)d_in[4];
    const float* bc1 = (const float*)d_in[5];
    const float* A1  = (const float*)d_in[6];
    const float* b1  = (const float*)d_in[7];
    const float* W2  = (const float*)d_in[8];
    const float* as2 = (const float*)d_in[9];
    const float* ad2 = (const float*)d_in[10];
    const float* bc2 = (const float*)d_in[11];
    const float* A2  = (const float*)d_in[12];
    const float* b2  = (const float*)d_in[13];
    const float* Hw1 = (const float*)d_in[14];
    const float* Hb1 = (const float*)d_in[15];
    const float* Hw2 = (const float*)d_in[16];
    const float* Hb2 = (const float*)d_in[17];
    const float* Hw3 = (const float*)d_in[18];
    const float* Hb3 = (const float*)d_in[19];
    const float* fcw = (const float*)d_in[20];
    const float* fcb = (const float*)d_in[21];
    float* out = (float*)d_out;

    const int DIN = 256, H1 = 512, H2 = 256;
    const int M  = in_sizes[0] / DIN;        // 50000 nodes
    const int Mp = (M + 63) & ~63;           // pad A-side buffers -> guard-free loads
    const int E  = in_sizes[1] / 2;          // 800000 edges

    // ---- workspace bump allocator (256B aligned) ----
    size_t off = 0;
    char* wsb = (char*)d_ws;
    auto alloc = [&](size_t bytes) -> void* {
        void* p = wsb + off;
        off = (off + bytes + 255) & ~(size_t)255;
        return p;
    };
    bf16* xbf  = (bf16*)alloc((size_t)Mp * DIN * 2);
    bf16* w1t  = (bf16*)alloc((size_t)DIN * H1 * 2);   // [H1][DIN]
    bf16* a1t  = (bf16*)alloc((size_t)DIN * H1 * 2);
    bf16* w2t  = (bf16*)alloc((size_t)H1 * H2 * 2);    // [H2][H1]
    bf16* a2t  = (bf16*)alloc((size_t)H1 * H2 * 2);
    bf16* hw1t = (bf16*)alloc((size_t)H2 * H2 * 2);
    bf16* hw2t = (bf16*)alloc((size_t)H2 * H2 * 2);
    bf16* hw3t = (bf16*)alloc((size_t)H2 * H2 * 2);
    bf16* h1bf = (bf16*)alloc((size_t)Mp * H1 * 2);
    float* aS  = (float*)alloc((size_t)M * 4);
    float* aD  = (float*)alloc((size_t)M * 4);
    float* mb  = (float*)alloc((size_t)M * 4);
    float* sb  = (float*)alloc((size_t)M * 4);
    float* lg  = (float*)alloc((size_t)E * 4);
    float* ewt = (float*)alloc((size_t)E * 4);
    float* agg = (float*)alloc((size_t)M * H1 * 4);    // reused (smaller) in phase 2
    bf16* act1bf = (bf16*)alloc((size_t)Mp * H1 * 2);
    bf16* h2bf   = (bf16*)alloc((size_t)Mp * H2 * 2);
    // reuse dead buffers:
    bf16* act2bf = xbf;                          // x dead after lin1 GEMM
    bf16* t1bf   = h1bf;                         // h1 dead after conv1 aggregate
    bf16* t2bf   = h1bf + (size_t)Mp * H2;
    bf16* t3bf   = h2bf;                         // h2 dead after conv2 aggregate

    auto cdiv = [](int a, int b) { return (a + b - 1) / b; };
    const int TB = 256;

    // ---- precision conversion (+ weight transpose to [N][K]) ----
    k_cvt_bf16  <<<4096, TB, 0, stream>>>(x,   xbf,  M * DIN);
    k_cvt_bf16_t<<<512,  TB, 0, stream>>>(W1,  w1t,  DIN, H1);
    k_cvt_bf16_t<<<512,  TB, 0, stream>>>(A1,  a1t,  DIN, H1);
    k_cvt_bf16_t<<<512,  TB, 0, stream>>>(W2,  w2t,  H1,  H2);
    k_cvt_bf16_t<<<512,  TB, 0, stream>>>(A2,  a2t,  H1,  H2);
    k_cvt_bf16_t<<<256,  TB, 0, stream>>>(Hw1, hw1t, H2,  H2);
    k_cvt_bf16_t<<<256,  TB, 0, stream>>>(Hw2, hw2t, H2,  H2);
    k_cvt_bf16_t<<<256,  TB, 0, stream>>>(Hw3, hw3t, H2,  H2);

    const int mB64 = Mp / 64;

    // ======== GAT layer 1 ========
    k_gemm_bf16<<<dim3(H1 / 128, mB64), TB, 0, stream>>>(
        xbf, w1t, nullptr, nullptr, nullptr, h1bf, M, H1, DIN, 0);
    k_attn_dots<<<cdiv(M * 32, TB), TB, 0, stream>>>(h1bf, as1, ad1, aS, aD, M, H1);
    // pre-seed scatter accumulator: agg = x @ A1 + b1 + b_conv1
    k_gemm_bf16<<<dim3(H1 / 128, mB64), TB, 0, stream>>>(
        xbf, a1t, b1, bc1, agg, nullptr, M, H1, DIN, 0);
    k_init_ms<<<cdiv(M, TB), TB, 0, stream>>>(mb, sb, M);
    k_edge_logits_max<<<cdiv(E, TB), TB, 0, stream>>>(ei, aS, aD, lg, mb, E, M);
    k_edge_exp_sum<<<cdiv(E, TB), TB, 0, stream>>>(ei, lg, mb, ewt, sb, E, M);
    k_edge_aggregate<<<cdiv(E * 32, TB), TB, 0, stream>>>(ei, ewt, sb, h1bf, agg, E, M, H1);
    k_relu_cvt<<<8192, TB, 0, stream>>>(agg, act1bf, M * H1, 1);

    // ======== GAT layer 2 ========
    k_gemm_bf16<<<dim3(H2 / 128, mB64), TB, 0, stream>>>(
        act1bf, w2t, nullptr, nullptr, nullptr, h2bf, M, H2, H1, 0);
    k_attn_dots<<<cdiv(M * 32, TB), TB, 0, stream>>>(h2bf, as2, ad2, aS, aD, M, H2);
    k_gemm_bf16<<<dim3(H2 / 128, mB64), TB, 0, stream>>>(
        act1bf, a2t, b2, bc2, agg, nullptr, M, H2, H1, 0);
    k_init_ms<<<cdiv(M, TB), TB, 0, stream>>>(mb, sb, M);
    k_edge_logits_max<<<cdiv(E, TB), TB, 0, stream>>>(ei, aS, aD, lg, mb, E, M);
    k_edge_exp_sum<<<cdiv(E, TB), TB, 0, stream>>>(ei, lg, mb, ewt, sb, E, M);
    k_edge_aggregate<<<cdiv(E * 32, TB), TB, 0, stream>>>(ei, ewt, sb, h2bf, agg, E, M, H2);
    k_relu_cvt<<<4096, TB, 0, stream>>>(agg, act2bf, M * H2, 0);  // no relu pre-MLP

    // ======== MLP (bias+relu fused into WMMA epilogue) ========
    k_gemm_bf16<<<dim3(H2 / 128, mB64), TB, 0, stream>>>(
        act2bf, hw1t, Hb1, nullptr, nullptr, t1bf, M, H2, H2, 1);
    k_gemm_bf16<<<dim3(H2 / 128, mB64), TB, 0, stream>>>(
        t1bf, hw2t, Hb2, nullptr, nullptr, t2bf, M, H2, H2, 1);
    k_gemm_bf16<<<dim3(H2 / 128, mB64), TB, 0, stream>>>(
        t2bf, hw3t, Hb3, nullptr, nullptr, t3bf, M, H2, H2, 1);

    // ======== classifier + softmax ========
    k_classify<<<cdiv(M * 32, TB), TB, 0, stream>>>(t3bf, fcw, fcb, out, M);
}
